// RecurGraphAgent_10548439679015
// MI455X (gfx1250) — matrix operand order repaired and checked by reading, not compiled, not run on hardware
//
#include <hip/hip_runtime.h>
#include <hip/hip_bf16.h>

#define Nn   20000
#define Ee   320000
#define Gg   64
#define FN   16
#define FE   8
#define CONV 32
#define LSTM 32

typedef __attribute__((ext_vector_type(16))) __bf16 v16bf;
typedef __attribute__((ext_vector_type(8)))  float  v8f;

// ---------- helpers ----------
__device__ __forceinline__ unsigned short f2bf(float f) {
  unsigned u = __float_as_uint(f);
  u += 0x7fffu + ((u >> 16) & 1u);          // RNE
  return (unsigned short)(u >> 16);
}
__device__ __forceinline__ unsigned pk2(float a, float b) {
  return (unsigned)f2bf(a) | ((unsigned)f2bf(b) << 16);
}
// Store 8 consecutive K values (k0 % 8 == 0) of A-row m into the LDS tile that
// is laid out in wave32 WMMA A-fragment order:
//   fragment row index = kstep*32 + (m + 16*half_c), 16 bf16 elements per row,
//   element j <-> kk where  kk = 8*half_c + (j&7) + (j>=8 ? 16 : 0)
__device__ __forceinline__ void store8(unsigned short* Z, int m, int k0, const float* v) {
  int s  = k0 >> 5, kk = k0 & 31;
  int hc = (kk >> 3) & 1;
  int j0 = (kk & 16) ? 8 : 0;
  unsigned* p = (unsigned*)&Z[(s * 32 + m + 16 * hc) * 16 + j0];
  p[0] = pk2(v[0], v[1]); p[1] = pk2(v[2], v[3]);
  p[2] = pk2(v[4], v[5]); p[3] = pk2(v[6], v[7]);
}
union FragU { uint4 q[2]; v16bf v; };
__device__ __forceinline__ v16bf load_frag(const unsigned short* p) {
  FragU u;
  u.q[0] = *(const uint4*)p;
  u.q[1] = *(const uint4*)(p + 8);
  return u.v;
}
__device__ __forceinline__ float sigm(float x) { return 1.f / (1.f + __expf(-x)); }
__device__ __forceinline__ unsigned ordf(float f) {
  unsigned u = __float_as_uint(f);
  return (u & 0x80000000u) ? ~u : (u | 0x80000000u);
}
__device__ __forceinline__ float unordf(unsigned v) {
  return __uint_as_float((v & 0x80000000u) ? (v & 0x7fffffffu) : ~v);
}

// ---------- kernel 1: fused NNConv edge messages via bf16 WMMA ----------
// msg = [z | x_src | 0] (K=160, bf16) @ W2 (160x32, bf16), scatter-add to agg[dst].
__global__ void __launch_bounds__(256)
edge_msg_wmma(const float* __restrict__ x, const float* __restrict__ ea,
              const int* __restrict__ ei, const float* __restrict__ lw,
              const float* __restrict__ lb, float* __restrict__ agg, int ntiles) {
  __shared__ __align__(16) unsigned short Wsw[2 * 5 * 32 * 16];   // B frags, 10 KB
  __shared__ __align__(16) unsigned short Zsw[8][5 * 32 * 16];    // A staging, 40 KB

  // Build swizzled B (W2): k<128 -> lw[(i*32+o)*8+f] with k=i*8+f; 128..143 -> bias; else 0
  for (int idx = threadIdx.x; idx < 2 * 5 * 32 * 16; idx += 256) {
    int j = idx & 15, L = (idx >> 4) & 31, s = (idx >> 9) % 5, nt = idx / 2560;
    int n = L & 15, half = L >> 4;
    int k = s * 32 + 8 * half + (j & 7) + ((j & 8) ? 16 : 0);
    int o = nt * 16 + n;
    float val;
    if (k < 128)      val = lw[((k >> 3) * 32 + o) * 8 + (k & 7)];
    else if (k < 144) val = lb[(k - 128) * 32 + o];
    else              val = 0.f;
    Wsw[idx] = f2bf(val);
  }
  __syncthreads();

  const int wave = threadIdx.x >> 5, lane = threadIdx.x & 31;
  const int m = lane & 15, hp = lane >> 4;
  unsigned short* Z = Zsw[wave];
  const int stride = gridDim.x * 8;

  for (int tile = blockIdx.x * 8 + wave; tile < ntiles; tile += stride) {
    int e = tile * 16 + m;
    int src = ei[e];
    float xv[8], eav[8];
#pragma unroll
    for (int t = 0; t < 8; t++) xv[t] = x[src * FN + hp * 8 + t];
#pragma unroll
    for (int f = 0; f < 8; f++) eav[f] = ea[e * FE + f];
    int enext = (tile + stride) * 16 + m;
    if (enext < Ee) __builtin_prefetch(&ea[enext * FE], 0, 0);

    // z[m][i*8+f] = x_src[i]*ea[f]; lanes split i by half (hp)
#pragma unroll
    for (int il = 0; il < 8; il++) {
      float row[8];
#pragma unroll
      for (int f = 0; f < 8; f++) row[f] = xv[il] * eav[f];
      store8(Z, m, (hp * 8 + il) * 8, row);
    }
    store8(Z, m, 128 + hp * 8, xv);               // bias carriers
    float zr[8] = {0, 0, 0, 0, 0, 0, 0, 0};
    store8(Z, m, 144 + hp * 8, zr);               // K padding

    asm volatile("s_wait_dscnt 0" ::: "memory");  // intra-wave LDS RAW fence

    v8f c0 = {}, c1 = {};
#pragma unroll
    for (int s = 0; s < 5; s++) {
      v16bf a  = load_frag(&Z[(s * 32 + lane) * 16]);
      v16bf b0 = load_frag(&Wsw[((0 * 5 + s) * 32 + lane) * 16]);
      v16bf b1 = load_frag(&Wsw[((1 * 5 + s) * 32 + lane) * 16]);
      c0 = __builtin_amdgcn_wmma_f32_16x16x32_bf16(false, a, false, b0, (short)0, c0, false, false);
      c1 = __builtin_amdgcn_wmma_f32_16x16x32_bf16(false, a, false, b1, (short)0, c1, false, false);
    }
    // C layout: lane = n + 16*hc, VGPR r -> row r + 8*hc
    int n = lane & 15, hc = lane >> 4;
#pragma unroll
    for (int r = 0; r < 8; r++) {
      int ee = tile * 16 + r + 8 * hc;
      int d = ei[Ee + ee];                        // dst
      atomicAdd(&agg[d * CONV + n],      c0[r]);
      atomicAdd(&agg[d * CONV + 16 + n], c1[r]);
    }
  }
}

// ---------- kernel 2: root/ReLU + fused LSTM gates via bf16 WMMA ----------
// gates = [h0 | h_prev] (K=64) @ [w_ih ; w_hh]^T-rearranged (64x128)
__global__ void __launch_bounds__(256)
node_lstm_wmma(const float* __restrict__ x, const float* __restrict__ initial,
               const float* __restrict__ agg, const float* __restrict__ root_w,
               const float* __restrict__ conv_bias,
               const float* __restrict__ w_ih, const float* __restrict__ w_hh,
               const float* __restrict__ b_ih, const float* __restrict__ b_hh,
               const float* __restrict__ hs_w, const float* __restrict__ hs_b,
               const float* __restrict__ cs_w, const float* __restrict__ cs_b,
               float* __restrict__ h_out, int ntiles) {
  __shared__ __align__(16) unsigned short Wsw[8 * 2 * 32 * 16];   // 16 KB
  __shared__ __align__(16) unsigned short Zsw[8][2 * 32 * 16];    // 16 KB

  for (int idx = threadIdx.x; idx < 8 * 2 * 32 * 16; idx += 256) {
    int j = idx & 15, L = (idx >> 4) & 31, s = (idx >> 9) & 1, nt = idx >> 10;
    int n = L & 15, half = L >> 4;
    int k = s * 32 + 8 * half + (j & 7) + ((j & 8) ? 16 : 0);
    int col = nt * 16 + n;
    float val = (k < 32) ? w_ih[col * 32 + k] : w_hh[col * 32 + (k - 32)];
    Wsw[idx] = f2bf(val);
  }
  __syncthreads();

  const int wave = threadIdx.x >> 5, lane = threadIdx.x & 31;
  const int m = lane & 15, hp = lane >> 4;
  unsigned short* Z = Zsw[wave];
  const int stride = gridDim.x * 8;

  for (int tile = blockIdx.x * 8 + wave; tile < ntiles; tile += stride) {
    int node = tile * 16 + m;
    float xr[16];
#pragma unroll
    for (int i = 0; i < 16; i++) xr[i] = x[node * FN + i];
    float init_v = initial[node];
    float h0v[16], hpv[16];
#pragma unroll
    for (int oo = 0; oo < 16; oo++) {
      int o = hp * 16 + oo;
      float acc = conv_bias[o] + agg[node * CONV + o];
#pragma unroll
      for (int i = 0; i < 16; i++) acc += xr[i] * root_w[i * CONV + o];
      h0v[oo] = fmaxf(acc, 0.f);                  // ReLU
      hpv[oo] = init_v * hs_w[o] + hs_b[o];       // h_prev
    }
    store8(Z, m, hp * 16,          h0v);
    store8(Z, m, hp * 16 + 8,      h0v + 8);
    store8(Z, m, 32 + hp * 16,     hpv);
    store8(Z, m, 32 + hp * 16 + 8, hpv + 8);
    asm volatile("s_wait_dscnt 0" ::: "memory");

    v8f c[8];
#pragma unroll
    for (int nt = 0; nt < 8; nt++) {
      v8f acc = {};
#pragma unroll
      for (int s = 0; s < 2; s++) {
        v16bf a = load_frag(&Z[(s * 32 + lane) * 16]);
        v16bf b = load_frag(&Wsw[((nt * 2 + s) * 32 + lane) * 16]);
        acc = __builtin_amdgcn_wmma_f32_16x16x32_bf16(false, a, false, b, (short)0, acc, false, false);
      }
      c[nt] = acc;
    }
    // gate combine: cols 0..31=i, 32..63=f, 64..95=g, 96..127=o; all 4 gate
    // values for a given (row m, unit j) land in the same lane.
    int n = lane & 15, hc = lane >> 4;
#pragma unroll
    for (int jh = 0; jh < 2; jh++) {
      int j = jh * 16 + n;
      float bi  = b_ih[j]      + b_hh[j];
      float bf_ = b_ih[32 + j] + b_hh[32 + j];
      float bg  = b_ih[64 + j] + b_hh[64 + j];
      float bo  = b_ih[96 + j] + b_hh[96 + j];
      float cw = cs_w[j], cb = cs_b[j];
#pragma unroll
      for (int r = 0; r < 8; r++) {
        int node2 = tile * 16 + r + 8 * hc;
        float cprev = initial[node2] * cw + cb;
        float gi = c[0 + jh][r] + bi;
        float gf = c[2 + jh][r] + bf_;
        float gg = c[4 + jh][r] + bg;
        float go = c[6 + jh][r] + bo;
        float cc = sigm(gf) * cprev + sigm(gi) * tanhf(gg);
        h_out[node2 * LSTM + j] = sigm(go) * tanhf(cc);
      }
    }
  }
}

// ---------- kernel 3a: attention gate MLP + node-head logits + maxes ----------
__global__ void gate_score_kernel(const float* __restrict__ h, const int* __restrict__ batch,
                                  const float* __restrict__ gw1, const float* __restrict__ gb1,
                                  const float* __restrict__ gw2, const float* __restrict__ gb2,
                                  const float* __restrict__ fn_w, const float* __restrict__ fn_b,
                                  float* __restrict__ score, float* __restrict__ logn,
                                  unsigned* __restrict__ segmax, unsigned* __restrict__ nmax) {
  int nd = blockIdx.x * blockDim.x + threadIdx.x;
  if (nd >= Nn) return;
  const float* hr = h + nd * LSTM;
  float hl[32];
#pragma unroll
  for (int k = 0; k < 32; k++) hl[k] = hr[k];
  float sc = gb2[0];
  for (int q = 0; q < 64; q++) {
    float acc = gb1[q];
#pragma unroll
    for (int k = 0; k < 32; k++) acc += gw1[q * 32 + k] * hl[k];
    sc += gw2[q] * fmaxf(acc, 0.f);
  }
  score[nd] = sc;
  atomicMax(&segmax[batch[nd]], ordf(sc));
  float lmax = -3.4e38f;
#pragma unroll
  for (int j = 0; j < 4; j++) {
    float lg = fn_b[j];
#pragma unroll
    for (int k = 0; k < 32; k++) lg += fn_w[j * 32 + k] * hl[k];
    logn[nd * 4 + j] = lg;
    lmax = fmaxf(lmax, lg);
  }
  atomicMax(nmax, ordf(lmax));
}

// ---------- kernel 3b: exponentials + denominators ----------
__global__ void exp_kernel(const int* __restrict__ batch, const float* __restrict__ score,
                           float* __restrict__ esc, float* __restrict__ logn,
                           const unsigned* __restrict__ segmax, float* __restrict__ segden,
                           const unsigned* __restrict__ nmax, float* __restrict__ nsum) {
  int nd = blockIdx.x * blockDim.x + threadIdx.x;
  if (nd >= Nn) return;
  int g = batch[nd];
  float ev = __expf(score[nd] - unordf(segmax[g]));
  esc[nd] = ev;
  atomicAdd(&segden[g], ev);
  float M = unordf(*nmax);
  float s = 0.f;
#pragma unroll
  for (int j = 0; j < 4; j++) {
    float e = __expf(logn[nd * 4 + j] - M);
    logn[nd * 4 + j] = e;
    s += e;
  }
  atomicAdd(nsum, s);
}

// ---------- kernel 3c: attention pooling + node softmax output ----------
__global__ void pool_out_kernel(const int* __restrict__ batch, const float* __restrict__ h,
                                const float* __restrict__ esc, const float* __restrict__ segden,
                                const float* __restrict__ logn, const float* __restrict__ nsum,
                                float* __restrict__ pooled, float* __restrict__ out_node) {
  int nd = blockIdx.x * blockDim.x + threadIdx.x;
  if (nd >= Nn) return;
  int g = batch[nd];
  float attn = esc[nd] / segden[g];
#pragma unroll
  for (int j = 0; j < 32; j++) atomicAdd(&pooled[g * 32 + j], h[nd * 32 + j] * attn);
  float inv = 1.f / nsum[0];
#pragma unroll
  for (int j = 0; j < 4; j++) out_node[nd * 4 + j] = logn[nd * 4 + j] * inv;
}

// ---------- kernel 4: graph head + 256-wide softmax (single block) ----------
__global__ void graph_head_kernel(const float* __restrict__ pooled, const float* __restrict__ fg_w,
                                  const float* __restrict__ fg_b, float* __restrict__ outg) {
  __shared__ float red[256];
  int t = threadIdx.x;
  int g = t >> 2, q = t & 3;
  float lg = fg_b[q];
#pragma unroll
  for (int k = 0; k < 32; k++) lg += fg_w[q * 32 + k] * pooled[g * 32 + k];
  red[t] = lg; __syncthreads();
  for (int s = 128; s > 0; s >>= 1) { if (t < s) red[t] = fmaxf(red[t], red[t + s]); __syncthreads(); }
  float M = red[0]; __syncthreads();
  float ev = __expf(lg - M);
  red[t] = ev; __syncthreads();
  for (int s = 128; s > 0; s >>= 1) { if (t < s) red[t] += red[t + s]; __syncthreads(); }
  outg[t] = ev / red[0];
}

// ---------- launcher ----------
extern "C" void kernel_launch(void* const* d_in, const int* in_sizes, int n_in,
                              void* d_out, int out_size, void* d_ws, size_t ws_size,
                              hipStream_t stream) {
  const float* x          = (const float*)d_in[0];
  const float* edge_attr  = (const float*)d_in[1];
  const float* initial    = (const float*)d_in[2];
  const int*   edge_index = (const int*)d_in[3];
  const int*   batch      = (const int*)d_in[4];
  const float* conv_lin_w = (const float*)d_in[5];
  const float* conv_lin_b = (const float*)d_in[6];
  const float* conv_root_w= (const float*)d_in[7];
  const float* conv_bias  = (const float*)d_in[8];
  const float* w_ih       = (const float*)d_in[9];
  const float* w_hh       = (const float*)d_in[10];
  const float* b_ih       = (const float*)d_in[11];
  const float* b_hh       = (const float*)d_in[12];
  const float* hs_w       = (const float*)d_in[13];
  const float* hs_b       = (const float*)d_in[14];
  const float* cs_w       = (const float*)d_in[15];
  const float* cs_b       = (const float*)d_in[16];
  const float* gate_w1    = (const float*)d_in[17];
  const float* gate_b1    = (const float*)d_in[18];
  const float* gate_w2    = (const float*)d_in[19];
  const float* gate_b2    = (const float*)d_in[20];
  const float* fg_w       = (const float*)d_in[21];
  const float* fg_b       = (const float*)d_in[22];
  const float* fn_w       = (const float*)d_in[23];
  const float* fn_b       = (const float*)d_in[24];
  float* out = (float*)d_out;

  float* ws = (float*)d_ws;
  float*    agg    = ws;                         // N*32
  float*    h      = ws + 640000;                // N*32
  float*    score  = ws + 1280000;               // N
  float*    esc    = ws + 1300000;               // N
  float*    logn   = ws + 1320000;               // N*4
  unsigned* segmax = (unsigned*)(ws + 1400000);  // G (ordered-float)
  float*    segden = ws + 1400064;               // G
  float*    pooled = ws + 1400128;               // G*32
  unsigned* nmax   = (unsigned*)(ws + 1402176);  // 1 (ordered-float)
  float*    nsum   = ws + 1402177;               // 1

  hipMemsetAsync(agg, 0, 640000 * sizeof(float), stream);
  hipMemsetAsync(ws + 1400000, 0, 2178 * sizeof(float), stream);  // maxes/dens/pooled/sums

  edge_msg_wmma<<<512, 256, 0, stream>>>(x, edge_attr, edge_index, conv_lin_w, conv_lin_b,
                                         agg, Ee / 16);
  node_lstm_wmma<<<160, 256, 0, stream>>>(x, initial, agg, conv_root_w, conv_bias,
                                          w_ih, w_hh, b_ih, b_hh, hs_w, hs_b, cs_w, cs_b,
                                          h, Nn / 16);
  int nb = (Nn + 255) / 256;
  gate_score_kernel<<<nb, 256, 0, stream>>>(h, batch, gate_w1, gate_b1, gate_w2, gate_b2,
                                            fn_w, fn_b, score, logn, segmax, nmax);
  exp_kernel<<<nb, 256, 0, stream>>>(batch, score, esc, logn, segmax, segden, nmax, nsum);
  pool_out_kernel<<<nb, 256, 0, stream>>>(batch, h, esc, segden, logn, nsum, pooled, out);
  graph_head_kernel<<<1, 256, 0, stream>>>(pooled, fg_w, fg_b, out + Nn * 4);
}
